// AttentionADCell_10256381903526
// MI455X (gfx1250) — matrix operand order repaired and checked by reading, not compiled, not run on hardware
//
#include <hip/hip_runtime.h>
#include <hip/hip_bf16.h>
#include <math.h>

// ---------------------------------------------------------------------------
// AttentionADCell fused kernel for gfx1250 (MI455X), wave32 + WMMA bf16.
//   D_IN=512, D_H=256, D_G=128, H=8, B=4096
//   N_total = 256 (Wh) + 2048 (Wz1) + 1024 (Wz2) = 3328 columns
// ---------------------------------------------------------------------------

#define D_IN   512
#define D_H    256
#define D_G    128
#define NHEAD  8
#define BATCH  4096
#define N_TOT  3328              // 256 + 2048 + 1024
#define NT     (N_TOT / 16)      // 208 N-tiles of 16 columns
#define KT     (D_IN / 32)       // 16 K-steps of 32
#define MT     (BATCH / 16)      // 256 batch tiles of 16 rows
#define EPS_LN  1e-6f
#define EPS_STD 1e-6f

typedef __attribute__((ext_vector_type(16))) __bf16 v16bf;
typedef __attribute__((ext_vector_type(8)))  float  v8f;

// round-to-nearest-even fp32 -> bf16 bits
__device__ __forceinline__ unsigned short f2bf(float f) {
    union { float f; unsigned u; } v; v.f = f;
    unsigned u = v.u;
    u += 0x7fffu + ((u >> 16) & 1u);
    return (unsigned short)(u >> 16);
}

__device__ __forceinline__ float wave_sum(float x) {
    #pragma unroll
    for (int m = 16; m >= 1; m >>= 1) x += __shfl_xor(x, m, 32);
    return x;
}

// ---------------------------------------------------------------------------
// Pack x (B x 512 fp32) into bf16 A-fragments, ISA per-lane order.
// Block (mt,kt) = 32 lanes * 16 bf16 (one v16bf / lane, 32B).
//   lane<16 : M=lane,   K = {0..7} U {16..23}   (vgpr 0..3 / 4..7, pairs)
//   lane>=16: M=lane-16,K = {8..15} U {24..31}
// ---------------------------------------------------------------------------
__global__ void pack_A_bf16(const float* __restrict__ x,
                            unsigned short* __restrict__ xb) {
    int o = blockIdx.x * blockDim.x + threadIdx.x;
    if (o >= BATCH * D_IN) return;
    int blk  = o >> 9;          // / 512 elements per (mt,kt) block
    int w    = o & 511;
    int lane = w >> 4;
    int e    = w & 15;
    int v    = e >> 1;
    int p    = e & 1;
    int mt   = blk / KT;
    int kt   = blk % KT;
    int row  = mt * 16 + (lane & 15);
    int koff = ((v >= 4) ? 16 : 0) + ((lane >= 16) ? 8 : 0) + ((v & 3) << 1) + p;
    int k    = kt * 32 + koff;
    xb[o] = f2bf(x[row * D_IN + k]);
}

// ---------------------------------------------------------------------------
// Pack [Wh | Wz1 | Wz2] (each (K=512, N) row-major) into bf16 B-fragments.
// Block (nt,kt): lane<16 -> col = nt*16+lane, K = kt*32 + {0..7} U {16..23} (pairs);
//                lane>=16 -> same col (lane-16), K = kt*32 + {8..15} U {24..31}.
// ---------------------------------------------------------------------------
__global__ void pack_B_bf16(const float* __restrict__ Wh,
                            const float* __restrict__ Wz1,
                            const float* __restrict__ Wz2,
                            unsigned short* __restrict__ Wt) {
    int o = blockIdx.x * blockDim.x + threadIdx.x;
    if (o >= D_IN * N_TOT) return;
    int blk  = o >> 9;
    int w    = o & 511;
    int lane = w >> 4;
    int e    = w & 15;
    int v    = e >> 1;
    int p    = e & 1;
    int nt   = blk / KT;
    int kt   = blk % KT;
    int col  = nt * 16 + (lane & 15);
    int koff = ((v >= 4) ? 16 : 0) + ((lane >= 16) ? 8 : 0) + ((v & 3) << 1) + p;
    int k    = kt * 32 + koff;
    float src;
    if (col < 256)        src = Wh [k * 256  + col];
    else if (col < 2304)  src = Wz1[k * 2048 + (col - 256)];
    else                  src = Wz2[k * 1024 + (col - 2304)];
    Wt[o] = f2bf(src);
}

// ---------------------------------------------------------------------------
// Fused main kernel: one workgroup = 16 batch rows, 256 threads = 8 waves.
// Stage 1: Z(16 x 3328) = Xtile @ [Wh|Wz1|Wz2] via v_wmma_f32_16x16x32_bf16.
//          Each wave handles 26 N-tiles as 13 PAIRS (nt, nt+8) with two
//          independent accumulators: one A-fragment load feeds two WMMAs,
//          and the two WMMA chains are independent (hides XDL accumulate
//          latency that a single chained accumulator would expose).
// Stage 2a: h = LN(relu(Z[:,0:256] + bh)) -> LDS + global h output.
// Stage 2b: per (row,g): w_row = sum_h z2[h,g]*z1[h,:]; standardize; tanh;
//           y[row,g] = w_row . h_row  (w never touches HBM: saves 512 MB).
// ---------------------------------------------------------------------------
__global__ void attn_ad_cell_fused(const unsigned short* __restrict__ xb,
                                   const unsigned short* __restrict__ Wt,
                                   const float* __restrict__ bh,
                                   const float* __restrict__ ln_scale,
                                   const float* __restrict__ ln_bias,
                                   float* __restrict__ out_h,
                                   float* __restrict__ out_y) {
    extern __shared__ float smem[];
    float* Zs = smem;                    // 16 * 3328 floats
    float* Hs = smem + 16 * N_TOT;       // 16 * 256 floats

    const int mt   = blockIdx.x;         // batch tile
    const int b0   = mt * 16;
    const int wave = threadIdx.x >> 5;
    const int lane = threadIdx.x & 31;

    // ---------------- Stage 1: WMMA GEMM (dual-accumulator N blocking) -----
    const v16bf* Ax = reinterpret_cast<const v16bf*>(xb) + (size_t)mt * KT * 32 + lane;
    const v16bf* Bw = reinterpret_cast<const v16bf*>(Wt);

    #pragma unroll 1
    for (int i = 0; i < 13; ++i) {
        const int nt0 = wave + 16 * i;      // tiles wave+8j, paired (2i, 2i+1)
        const int nt1 = nt0 + 8;
        const v16bf* B0 = Bw + (size_t)nt0 * KT * 32 + lane;
        const v16bf* B1 = Bw + (size_t)nt1 * KT * 32 + lane;
        v8f acc0 = {};
        v8f acc1 = {};
        #pragma unroll 2
        for (int kt = 0; kt < KT; ++kt) {
            v16bf a  = Ax[kt * 32];
            v16bf b0 = B0[kt * 32];
            v16bf b1 = B1[kt * 32];
            acc0 = __builtin_amdgcn_wmma_f32_16x16x32_bf16(
                false, a, false, b0, (short)0, acc0, false, false);
            acc1 = __builtin_amdgcn_wmma_f32_16x16x32_bf16(
                false, a, false, b1, (short)0, acc1, false, false);
        }
        // C/D layout: vgpr r -> M = r (lanes 0-15) / 8+r (lanes 16-31); N = lane%16
        const int n0    = nt0 * 16 + (lane & 15);
        const int n1    = nt1 * 16 + (lane & 15);
        const int rbase = (lane < 16) ? 0 : 8;
        #pragma unroll
        for (int r = 0; r < 8; ++r) {
            Zs[(rbase + r) * N_TOT + n0] = acc0[r];
            Zs[(rbase + r) * N_TOT + n1] = acc1[r];
        }
    }
    __syncthreads();

    // ---------------- Stage 2a: h = LN(relu(z + bh)) ----------------
    for (int r = wave; r < 16; r += 8) {
        float v[8];
        float s = 0.f, sq = 0.f;
        #pragma unroll
        for (int i = 0; i < 8; ++i) {
            int n = lane + 32 * i;
            float t = Zs[r * N_TOT + n] + bh[n];
            t = fmaxf(t, 0.f);
            v[i] = t;
            s += t; sq += t * t;
        }
        s  = wave_sum(s);
        sq = wave_sum(sq);
        float mu  = s * (1.f / D_H);
        float var = sq * (1.f / D_H) - mu * mu;
        float rs  = rsqrtf(fmaxf(var, 0.f) + EPS_LN);
        #pragma unroll
        for (int i = 0; i < 8; ++i) {
            int n = lane + 32 * i;
            float hn = (v[i] - mu) * rs * ln_scale[n] + ln_bias[n];
            Hs[r * D_H + n] = hn;
            out_h[(b0 + r) * D_H + n] = hn;
        }
    }
    __syncthreads();

    // ---------------- Stage 2b: w rows, standardize, tanh, y ----------------
    // 16 rows * 128 g = 2048 tasks; each wave handles 256. Lane owns 8
    // contiguous d values (32B float4 LDS loads).
    for (int t = wave; t < 16 * D_G; t += 8) {
        int r = t >> 7;
        int g = t & (D_G - 1);
        const float* zrow = &Zs[r * N_TOT];

        float sc[NHEAD];
        #pragma unroll
        for (int h = 0; h < NHEAD; ++h)
            sc[h] = zrow[2304 + h * D_G + g];      // z2[b][h][g] (LDS broadcast)

        float wv[8];
        #pragma unroll
        for (int j = 0; j < 8; ++j) wv[j] = 0.f;

        const float* z1p = zrow + 256;              // z1[b][h][d]
        int d0 = lane * 8;
        #pragma unroll
        for (int h = 0; h < NHEAD; ++h) {
            const float4 a = *reinterpret_cast<const float4*>(&z1p[h * D_H + d0]);
            const float4 c = *reinterpret_cast<const float4*>(&z1p[h * D_H + d0 + 4]);
            float s = sc[h];
            wv[0] += s * a.x; wv[1] += s * a.y; wv[2] += s * a.z; wv[3] += s * a.w;
            wv[4] += s * c.x; wv[5] += s * c.y; wv[6] += s * c.z; wv[7] += s * c.w;
        }

        float s = 0.f, sq = 0.f;
        #pragma unroll
        for (int j = 0; j < 8; ++j) { s += wv[j]; sq += wv[j] * wv[j]; }
        s  = wave_sum(s);
        sq = wave_sum(sq);
        float mu  = s * (1.f / D_H);
        float var = sq * (1.f / D_H) - mu * mu;
        float sd  = sqrtf(fmaxf(var, 0.f));
        float inv = 1.f / (sd + EPS_STD);

        const float4 h0 = *reinterpret_cast<const float4*>(&Hs[r * D_H + d0]);
        const float4 h1 = *reinterpret_cast<const float4*>(&Hs[r * D_H + d0 + 4]);
        float hv[8] = { h0.x, h0.y, h0.z, h0.w, h1.x, h1.y, h1.z, h1.w };

        float yp = 0.f;
        #pragma unroll
        for (int j = 0; j < 8; ++j)
            yp += tanhf((wv[j] - mu) * inv) * hv[j];
        yp = wave_sum(yp);
        if (lane == 0)
            out_y[(b0 + r) * D_G + g] = yp;
    }
}

// ---------------------------------------------------------------------------
extern "C" void kernel_launch(void* const* d_in, const int* in_sizes, int n_in,
                              void* d_out, int out_size, void* d_ws, size_t ws_size,
                              hipStream_t stream) {
    const float* x        = (const float*)d_in[0];   // (4096, 512)
    const float* Wh       = (const float*)d_in[1];   // (512, 256)
    const float* bh       = (const float*)d_in[2];   // (256,)
    const float* Wz1      = (const float*)d_in[3];   // (512, 2048)
    const float* Wz2      = (const float*)d_in[4];   // (512, 1024)
    const float* ln_scale = (const float*)d_in[5];   // (256,)
    const float* ln_bias  = (const float*)d_in[6];   // (256,)

    float* out_h = (float*)d_out;                     // (4096, 256)
    float* out_y = out_h + (size_t)BATCH * D_H;       // (4096, 128)

    // workspace: bf16 packed A then bf16 packed B
    unsigned short* xb = (unsigned short*)d_ws;                    // 4 MB
    unsigned short* Wt = xb + (size_t)BATCH * D_IN;                // 3.4 MB

    {
        int n = BATCH * D_IN;
        pack_A_bf16<<<(n + 255) / 256, 256, 0, stream>>>(x, xb);
    }
    {
        int n = D_IN * N_TOT;
        pack_B_bf16<<<(n + 255) / 256, 256, 0, stream>>>(Wh, Wz1, Wz2, Wt);
    }

    size_t smem = (size_t)(16 * N_TOT + 16 * D_H) * sizeof(float); // 224 KB < 320 KB WGP LDS
    attn_ad_cell_fused<<<MT, 256, smem, stream>>>(xb, Wt, bh, ln_scale, ln_bias,
                                                  out_h, out_y);
}
